// SimpleRNN_73366631350621
// MI455X (gfx1250) — compile-verified, matrix operand order checked by reading, not run on hardware
//
#include <hip/hip_runtime.h>

typedef __attribute__((ext_vector_type(16))) _Float16 v16h;
typedef __attribute__((ext_vector_type(8)))  _Float16 v8h;
typedef __attribute__((ext_vector_type(4)))  _Float16 v4h;
typedef __attribute__((ext_vector_type(8)))  float    v8f;

// Model dims
#define BB   128
#define TT   512
#define EE   512
#define HH   512
#define FFN  1024
#define NCLS 10
#define BT   (BB*TT)

// ---------------------------------------------------------------------------
// A-fragment loader: 16x32 f16 A tile, natural row-major source.
// lane 0-15 -> rows 0..15 (K lower halves), lane 16-31 -> same rows, K upper.
// Per documented layout: element e: K = (e<8) ? half*8+e : 16+half*8+(e-8)
// -> two contiguous 8-half (16B) loads per lane.
// ---------------------------------------------------------------------------
__device__ __forceinline__ v16h load_a_frag(const _Float16* __restrict__ A,
                                            int row, int ldk, int kbase, int half) {
  const _Float16* p = A + (size_t)row * ldk + kbase + half * 8;
  v8h lo = *(const v8h*)(p);
  v8h hi = *(const v8h*)(p + 16);
  return __builtin_shufflevector(lo, hi, 0,1,2,3,4,5,6,7,8,9,10,11,12,13,14,15);
}

// ---------------------------------------------------------------------------
// Weight pre-swizzle: f32 [K, srcN] row-major -> f16 fragments laid out so a
// wave reads its 32x16 B tile as contiguous 16 halves per lane.
// dst[((kt*nnTiles+nt)*32 + L)*16 + e] = W[kt*32 + K(e,L>>4), nt*16 + (L&15)]
// ---------------------------------------------------------------------------
__global__ void convert_weight_swz(const float* __restrict__ src,
                                   _Float16* __restrict__ dst,
                                   int K, int srcN, int Np) {
  int tid = blockIdx.x * blockDim.x + threadIdx.x;
  int total = K * Np;
  if (tid >= total) return;
  int e    = tid & 15;
  int L    = (tid >> 4) & 31;
  int frag = tid >> 9;                 // 32*16 = 512 elements per fragment
  int nnTiles = Np >> 4;
  int nt = frag % nnTiles;
  int kt = frag / nnTiles;
  int h  = L >> 4;
  int n  = nt * 16 + (L & 15);
  int kk = kt * 32 + ((e < 8) ? (h * 8 + e) : (16 + h * 8 + (e - 8)));
  float v = (n < srcN) ? src[(size_t)kk * srcN + n] : 0.0f;
  dst[tid] = (_Float16)v;
}

// ---------------------------------------------------------------------------
// Embedding gather + f32->f16 convert. 4 elements per thread.
// ---------------------------------------------------------------------------
__global__ void gather_embed(const int* __restrict__ x,
                             const float* __restrict__ table,
                             _Float16* __restrict__ out, int rows) {
  int tid = blockIdx.x * blockDim.x + threadIdx.x;
  int row = tid >> 7;                  // EE/4 = 128 quads per row
  if (row >= rows) return;
  int e4  = (tid & 127) * 4;
  int tok = x[row];
  float4 v = *(const float4*)(table + (size_t)tok * EE + e4);
  v4h o;
  o[0] = (_Float16)v.x; o[1] = (_Float16)v.y;
  o[2] = (_Float16)v.z; o[3] = (_Float16)v.w;
  *(v4h*)(out + (size_t)row * EE + e4) = o;
}

// ---------------------------------------------------------------------------
// Generic WMMA GEMM: C[M,N] = A[M,K](f16, row-major) x Bswz[K,N] + bias
// Block = 256 threads = 8 waves (4 M-waves x 2 N-waves) -> 64x128 block tile.
// Each wave: 1 M-tile x 4 N-tiles, A-fragment reused across 4 WMMAs.
// ---------------------------------------------------------------------------
template<bool RELU, bool STORE32, bool STORE16>
__global__ __launch_bounds__(256)
void gemm_wmma(const _Float16* __restrict__ A, const _Float16* __restrict__ Bs,
               const float* __restrict__ bias, float* __restrict__ out32,
               _Float16* __restrict__ out16, int M, int N, int K) {
  const int lane  = threadIdx.x & 31;
  const int wave  = threadIdx.x >> 5;
  const int half  = lane >> 4;
  const int ln16  = lane & 15;
  const int waveM = wave >> 1;
  const int waveN = wave & 1;
  const int nnTiles = N >> 4;
  const int rowBase = blockIdx.y * 64 + waveM * 16;
  const int ntBase  = blockIdx.x * 8 + waveN * 4;

  v8f acc[4] = {};
  const int nkt = K >> 5;
  for (int kt = 0; kt < nkt; ++kt) {
    v16h a = load_a_frag(A, rowBase + ln16, K, kt * 32, half);
#pragma unroll
    for (int j = 0; j < 4; ++j) {
      const v16h b = *(const v16h*)(Bs +
          (((size_t)(kt * nnTiles + ntBase + j)) * 32 + lane) * 16);
      acc[j] = __builtin_amdgcn_wmma_f32_16x16x32_f16(false, a, false, b,
                                                      (short)0, acc[j], false, false);
    }
  }
#pragma unroll
  for (int j = 0; j < 4; ++j) {
    int col = (ntBase + j) * 16 + ln16;
    float bv = bias ? bias[col] : 0.0f;
#pragma unroll
    for (int r = 0; r < 8; ++r) {
      int row = rowBase + half * 8 + r;
      float v = acc[j][r] + bv;
      if (RELU) v = v > 0.0f ? v : 0.0f;
      size_t idx = (size_t)row * N + col;
      if (STORE32) out32[idx] = v;
      if (STORE16) out16[idx] = (_Float16)v;
    }
  }
}

// ---------------------------------------------------------------------------
// One RNN timestep: h_new = mask ? relu(xW[:,t,:] + h_old @ W_rec) : h_old
// M=128 (batch), N=K=512. Writes next-state (ping-pong) and h_seq[:,t,:].
// ---------------------------------------------------------------------------
__global__ __launch_bounds__(256)
void rnn_step(const _Float16* __restrict__ hIn, const _Float16* __restrict__ Ws,
              const float* __restrict__ xW, const int* __restrict__ x,
              _Float16* __restrict__ hOut, _Float16* __restrict__ hSeq, int t) {
  const int lane  = threadIdx.x & 31;
  const int wave  = threadIdx.x >> 5;
  const int half  = lane >> 4;
  const int ln16  = lane & 15;
  const int waveM = wave >> 1;
  const int waveN = wave & 1;
  const int nnTiles = HH >> 4;
  const int rowBase = blockIdx.y * 64 + waveM * 16;
  const int ntBase  = blockIdx.x * 8 + waveN * 4;

  v8f acc[4] = {};
  for (int kt = 0; kt < (HH >> 5); ++kt) {
    v16h a = load_a_frag(hIn, rowBase + ln16, HH, kt * 32, half);
#pragma unroll
    for (int j = 0; j < 4; ++j) {
      const v16h b = *(const v16h*)(Ws +
          (((size_t)(kt * nnTiles + ntBase + j)) * 32 + lane) * 16);
      acc[j] = __builtin_amdgcn_wmma_f32_16x16x32_f16(false, a, false, b,
                                                      (short)0, acc[j], false, false);
    }
  }
#pragma unroll
  for (int j = 0; j < 4; ++j) {
    int col = (ntBase + j) * 16 + ln16;
#pragma unroll
    for (int r = 0; r < 8; ++r) {
      int b = rowBase + half * 8 + r;            // batch index
      int tok = x[b * TT + t];
      size_t seqIdx = ((size_t)(b * TT + t)) * HH + col;
      float v = acc[j][r] + xW[seqIdx];
      v = v > 0.0f ? v : 0.0f;
      _Float16 res = (tok != 0) ? (_Float16)v : hIn[(size_t)b * HH + col];
      hOut[(size_t)b * HH + col] = res;
      hSeq[seqIdx] = res;
    }
  }
}

// ---------------------------------------------------------------------------
// Head: logits = z2 @ Wo + bo (N padded 10->16 with zeros), fused softmax.
// One wave per 16-row tile; softmax over the 16-lane groups via shfl_xor.
// ---------------------------------------------------------------------------
__global__ __launch_bounds__(256)
void head_softmax(const _Float16* __restrict__ A, const _Float16* __restrict__ Ws,
                  const float* __restrict__ bo, float* __restrict__ out) {
  const int lane  = threadIdx.x & 31;
  const int wave  = threadIdx.x >> 5;
  const int half  = lane >> 4;
  const int ln16  = lane & 15;
  const int mtile = blockIdx.x * 8 + wave;
  const int rowBase = mtile * 16;

  v8f acc = {};
  for (int kt = 0; kt < (FFN >> 5); ++kt) {
    v16h a = load_a_frag(A, rowBase + ln16, FFN, kt * 32, half);
    const v16h b = *(const v16h*)(Ws + ((size_t)kt * 32 + lane) * 16); // nnTiles=1
    acc = __builtin_amdgcn_wmma_f32_16x16x32_f16(false, a, false, b,
                                                 (short)0, acc, false, false);
  }
  const int col = ln16;
  float bv = (col < NCLS) ? bo[col] : 0.0f;
#pragma unroll
  for (int r = 0; r < 8; ++r) {
    int row = rowBase + half * 8 + r;
    float logit = (col < NCLS) ? (acc[r] + bv) : -3.0e38f;
    float m = logit;
    for (int w = 8; w >= 1; w >>= 1) m = fmaxf(m, __shfl_xor(m, w, 16));
    float p = (col < NCLS) ? __expf(logit - m) : 0.0f;
    float s = p;
    for (int w = 8; w >= 1; w >>= 1) s += __shfl_xor(s, w, 16);
    if (col < NCLS) out[(size_t)row * NCLS + col] = p / s;
  }
}

// ---------------------------------------------------------------------------
// Host-side orchestration
// ---------------------------------------------------------------------------
extern "C" void kernel_launch(void* const* d_in, const int* in_sizes, int n_in,
                              void* d_out, int out_size, void* d_ws, size_t ws_size,
                              hipStream_t stream) {
  const int*   x         = (const int*)  d_in[0];
  const float* emb_table = (const float*)d_in[1];
  const float* W_in      = (const float*)d_in[2];
  const float* W_rec     = (const float*)d_in[3];
  const float* b_rnn     = (const float*)d_in[4];
  const float* W1        = (const float*)d_in[5];
  const float* b1        = (const float*)d_in[6];
  const float* W2        = (const float*)d_in[7];
  const float* b2        = (const float*)d_in[8];
  const float* Wo        = (const float*)d_in[9];
  const float* bo        = (const float*)d_in[10];
  float* out = (float*)d_out;

  char* ws = (char*)d_ws;
  size_t off = 0;
  auto alloc = [&](size_t bytes) -> void* {
    void* p = ws + off;
    off = (off + bytes + 255) & ~(size_t)255;
    return p;
  };

  _Float16* WinS  = (_Float16*)alloc((size_t)EE * HH * 2);
  _Float16* WrecS = (_Float16*)alloc((size_t)HH * HH * 2);
  _Float16* W1S   = (_Float16*)alloc((size_t)HH * FFN * 2);
  _Float16* W2S   = (_Float16*)alloc((size_t)FFN * FFN * 2);
  _Float16* WoS   = (_Float16*)alloc((size_t)FFN * 16 * 2);
  _Float16* emb16 = (_Float16*)alloc((size_t)BT * EE * 2);
  float*    xWbuf = (float*)   alloc((size_t)BT * HH * 4);   // aliased by z1 later
  _Float16* hSeq  = (_Float16*)alloc((size_t)BT * HH * 2);
  _Float16* hA    = (_Float16*)alloc((size_t)BB * HH * 2);
  _Float16* hB    = (_Float16*)alloc((size_t)BB * HH * 2);
  _Float16* z2buf = (_Float16*)alloc((size_t)BT * FFN * 2);
  _Float16* z1buf = (_Float16*)xWbuf;  // xW (f32) dead after recurrence; same byte size

  // 1. Pre-swizzle weights to f16 B-fragment layout
  convert_weight_swz<<<(EE * HH + 255) / 256, 256, 0, stream>>>(W_in,  WinS,  EE,  HH,   HH);
  convert_weight_swz<<<(HH * HH + 255) / 256, 256, 0, stream>>>(W_rec, WrecS, HH,  HH,   HH);
  convert_weight_swz<<<(HH * FFN + 255) / 256, 256, 0, stream>>>(W1,   W1S,   HH,  FFN,  FFN);
  convert_weight_swz<<<(FFN * FFN + 255) / 256, 256, 0, stream>>>(W2,  W2S,   FFN, FFN,  FFN);
  convert_weight_swz<<<(FFN * 16 + 255) / 256, 256, 0, stream>>>(Wo,   WoS,   FFN, NCLS, 16);

  // 2. Embedding gather -> f16
  gather_embed<<<(BT * (EE / 4) + 255) / 256, 256, 0, stream>>>(x, emb_table, emb16, BT);

  // 3. xW = emb @ W_in + b_rnn   (f32 out)
  gemm_wmma<false, true, false><<<dim3(HH / 128, BT / 64), 256, 0, stream>>>(
      emb16, WinS, b_rnn, xWbuf, (_Float16*)nullptr, BT, HH, EE);

  // 4. Recurrence: 512 sequential steps, ping-pong state
  (void)hipMemsetAsync(hA, 0, (size_t)BB * HH * 2, stream);
  for (int t = 0; t < TT; ++t) {
    const _Float16* hin = (t & 1) ? hB : hA;
    _Float16*       hout = (t & 1) ? hA : hB;
    rnn_step<<<dim3(HH / 128, BB / 64), 256, 0, stream>>>(hin, WrecS, xWbuf, x,
                                                          hout, hSeq, t);
  }

  // 5. z1 = relu(h_seq @ W1 + b1)   (f16 out, aliases xW buffer)
  gemm_wmma<true, false, true><<<dim3(FFN / 128, BT / 64), 256, 0, stream>>>(
      hSeq, W1S, b1, (float*)nullptr, z1buf, BT, FFN, HH);

  // 6. z2 = relu(z1 @ W2 + b2)      (f16 out)
  gemm_wmma<true, false, true><<<dim3(FFN / 128, BT / 64), 256, 0, stream>>>(
      z1buf, W2S, b2, (float*)nullptr, z2buf, BT, FFN, FFN);

  // 7. logits + softmax -> d_out
  head_softmax<<<BT / 16 / 8, 256, 0, stream>>>(z2buf, WoS, bo, out);
}
